// GCN_36532991820037
// MI455X (gfx1250) — matrix operand order repaired and checked by reading, not compiled, not run on hardware
//
#include <hip/hip_runtime.h>
#include <hip/hip_bf16.h>

// CDNA5 / gfx1250 wave32 WMMA implementation of the GCN forward pass.
// Dense layers -> v_wmma_f32_16x16x32_bf16 (bf16 A/B, f32 accumulate) with the
// B panel staged through LDS (double-buffered, cooperative 256-thread fetch) so
// WMMAs read operands via ds_load_b128 while the next panel's global loads are
// in flight. A is loaded fp32 direct-from-global, converted to bf16 in
// registers, pipelined one k-step ahead.
// SpMM -> coalesced float4 gather + global_atomic_add_f32 scatter (L2-resident).

typedef __attribute__((ext_vector_type(16))) __bf16 v16bf;
typedef __attribute__((ext_vector_type(8)))  float  v8f;

#define NFEAT_ 512
#define NHID_  256
#define NLAY_  4
#define EVO_   1024

#define BROW_   80                    // LDS bytes per B row (64B data + 16B pad)
#define BPANEL_ (256 * BROW_)         // 20 KB per panel buffer

// ---------------------------------------------------------------------------
// Transpose fp32 weight (K x Ncols, row-major, y = x@W) into bf16 N-major
// layout Wt[n][k] so the cooperative panel fetch is contiguous 64B per row.
// ---------------------------------------------------------------------------
__global__ __launch_bounds__(256) void k_transpose_bf16(
    const float* __restrict__ W, __bf16* __restrict__ Wt, int K, int Ncols)
{
    int idx = blockIdx.x * 256 + threadIdx.x;
    if (idx >= K * Ncols) return;
    int n = idx / K;
    int k = idx - n * K;
    Wt[idx] = (__bf16)W[(size_t)k * Ncols + n];
}

__global__ __launch_bounds__(256) void k_zero(float* __restrict__ p, size_t n)
{
    size_t i = (size_t)blockIdx.x * 256 + threadIdx.x;
    size_t stride = (size_t)gridDim.x * 256;
    for (; i < n; i += stride) p[i] = 0.0f;
}

// ---------------------------------------------------------------------------
// WMMA GEMM: C[M x 256] = act( A[M x K](fp32, lda) @ Wt^T + bias )
//   Wt: bf16, [256][K] N-major.
//   Block = 256 threads = 8 waves; each wave computes a 16 x 256 strip
//   (16 v8f accumulators). B panel (32 x 256 = 16 KB bf16 per k-step) is
//   ping-pong staged in LDS by all 256 threads; fragments come from
//   ds_load_b128 pairs. Next panel's global loads issue before the WMMA chain.
// ---------------------------------------------------------------------------
__global__ __launch_bounds__(256) void k_gemm_wmma(
    const float* __restrict__ A, int lda,
    const __bf16* __restrict__ Wt,
    const float* __restrict__ bias,   // nullable
    float* __restrict__ C, int ldc,
    int M, int K, int relu)
{
    __shared__ __align__(16) unsigned char ldsB[2 * BPANEL_];   // 40 KB

    const int tid  = threadIdx.x;
    const int wave = tid >> 5;
    const int lane = tid & 31;
    const int m0   = (blockIdx.x * 8 + wave) * 16;
    const bool active = (m0 < M);        // wave-uniform; inactive waves still
                                         // join cooperative loads + barriers

    const int half = lane >> 4;          // 0: lanes 0-15, 1: lanes 16-31
    const int l15  = lane & 15;

    v8f acc[16];
#pragma unroll
    for (int n = 0; n < 16; ++n) acc[n] = (v8f){0,0,0,0,0,0,0,0};

    // A fragment layout (contiguous k order used here, same as B):
    //   lanes 0-15 : row=l15, K = {0..15}
    //   lanes 16-31: row=l15, K = {16..31}
    const float* arow = A + (size_t)((active ? m0 : 0) + l15) * lda;
    const int kA0 = half * 16;

    // B fragment: lane = column (l15); lanes 0-15 K=0..15, lanes 16-31 K=16..31
    // LDS row r (= output column) at byte r*BROW_; fragment = 32B at half*32.
    const int fragBase = l15 * BROW_ + half * 32;

    // Cooperative panel fetch: thread t covers Wt row t (64 B per k-step).
    const __bf16* wrow = Wt + (size_t)tid * K;

    auto cvtA = [&](float4 a01, float4 a23, float4 a45, float4 a67) -> v16bf {
        v16bf af;
        af[0]  = (__bf16)a01.x; af[1]  = (__bf16)a01.y;
        af[2]  = (__bf16)a01.z; af[3]  = (__bf16)a01.w;
        af[4]  = (__bf16)a23.x; af[5]  = (__bf16)a23.y;
        af[6]  = (__bf16)a23.z; af[7]  = (__bf16)a23.w;
        af[8]  = (__bf16)a45.x; af[9]  = (__bf16)a45.y;
        af[10] = (__bf16)a45.z; af[11] = (__bf16)a45.w;
        af[12] = (__bf16)a67.x; af[13] = (__bf16)a67.y;
        af[14] = (__bf16)a67.z; af[15] = (__bf16)a67.w;
        return af;
    };

    // ---- prologue: panel 0 -> LDS buffer 0, A(k=0) -> registers ----
    {
        uint4 p0 = *(const uint4*)(wrow + 0);
        uint4 p1 = *(const uint4*)(wrow + 8);
        uint4 p2 = *(const uint4*)(wrow + 16);
        uint4 p3 = *(const uint4*)(wrow + 24);
        unsigned char* myRow = ldsB + tid * BROW_;
        *(uint4*)(myRow + 0)  = p0;
        *(uint4*)(myRow + 16) = p1;
        *(uint4*)(myRow + 32) = p2;
        *(uint4*)(myRow + 48) = p3;
    }
    v16bf af;
    if (active) {
        const float* ap0 = arow + kA0;
        float4 ra0 = *(const float4*)(ap0 + 0);
        float4 ra1 = *(const float4*)(ap0 + 4);
        float4 ra2 = *(const float4*)(ap0 + 8);
        float4 ra3 = *(const float4*)(ap0 + 12);
        af = cvtA(ra0, ra1, ra2, ra3);
    }
    __syncthreads();

    const int nsteps = K >> 5;
    for (int i = 0; i < nsteps; ++i) {
        const int  k    = i << 5;
        const bool more = (i + 1 < nsteps);
        unsigned char* cbuf = ldsB + (i & 1) * BPANEL_;

        // ---- issue next panel's global loads (land in regs during compute) --
        uint4 q0, q1, q2, q3;
        if (more) {
            const __bf16* wn = wrow + (k + 32);
            q0 = *(const uint4*)(wn + 0);
            q1 = *(const uint4*)(wn + 8);
            q2 = *(const uint4*)(wn + 16);
            q3 = *(const uint4*)(wn + 24);
        }
        // ---- issue next A loads; convert after the WMMA chain ----
        float4 na0, na1, na2, na3;
        if (active) {
            const float* apn = arow + (more ? k + 32 : 0) + kA0;
            na0 = *(const float4*)(apn + 0);
            na1 = *(const float4*)(apn + 4);
            na2 = *(const float4*)(apn + 8);
            na3 = *(const float4*)(apn + 12);
        }

        // ---- 16 WMMAs, B fragments from LDS (2 x ds_load_b128 each) ----
        if (active) {
#pragma unroll
            for (int n = 0; n < 16; ++n) {
                v16bf bf = *(const v16bf*)(cbuf + n * (16 * BROW_) + fragBase);
                acc[n] = __builtin_amdgcn_wmma_f32_16x16x32_bf16(
                    /*neg_a=*/false, af, /*neg_b=*/false, bf,
                    /*c_mod=*/(short)0, acc[n], /*reuse_a=*/false, /*reuse_b=*/false);
            }
        }

        // ---- publish panel i+1 into the other buffer ----
        if (more) {
            __syncthreads();   // everyone done reading the buffer we overwrite
            unsigned char* nrow = ldsB + ((i + 1) & 1) * BPANEL_ + tid * BROW_;
            *(uint4*)(nrow + 0)  = q0;
            *(uint4*)(nrow + 16) = q1;
            *(uint4*)(nrow + 32) = q2;
            *(uint4*)(nrow + 48) = q3;
            __syncthreads();   // panel i+1 visible to all waves
        }
        if (active) af = cvtA(na0, na1, na2, na3);
    }

    if (!active) return;

    // C/D layout: VGPR r -> row m0 + half*8 + r ; column = ntile*16 + l15
    const int mrow = m0 + half * 8;
#pragma unroll
    for (int n = 0; n < 16; ++n) {
        const int col = (n << 4) + l15;
        float bv = bias ? bias[col] : 0.0f;
#pragma unroll
        for (int r = 0; r < 8; ++r) {
            float v = acc[n][r] + bv;
            if (relu) v = v > 0.0f ? v : 0.0f;
            C[(size_t)(mrow + r) * ldc + col] = v;
        }
    }
}

// ---------------------------------------------------------------------------
// SpMM scatter: agg[dst] += w_e * h[src]  (one wave per edge, 8 feats/lane)
// Gather is fully coalesced (32 lanes cover one 1 KB row); scatter uses
// global_atomic_add_f32 which resolves in L2 (h and agg are L2-resident).
// ---------------------------------------------------------------------------
__global__ __launch_bounds__(256) void k_spmm_scatter(
    const float* __restrict__ h,
    const int* __restrict__ esrc, const int* __restrict__ edst,
    const float* __restrict__ ew,
    float* __restrict__ agg, int ldagg, int E)
{
    int t = blockIdx.x * 256 + threadIdx.x;
    int e = t >> 5;
    if (e >= E) return;
    int lane = t & 31;
    int s = esrc[e], d = edst[e];
    float w = ew[e];
    const float4* hp = (const float4*)(h + (size_t)s * NHID_) + (lane << 1);
    float4 v0 = hp[0], v1 = hp[1];
    float* ap = agg + (size_t)d * ldagg + (lane << 3);
    atomicAdd(ap + 0, w * v0.x);
    atomicAdd(ap + 1, w * v0.y);
    atomicAdd(ap + 2, w * v0.z);
    atomicAdd(ap + 3, w * v0.w);
    atomicAdd(ap + 4, w * v1.x);
    atomicAdd(ap + 5, w * v1.y);
    atomicAdd(ap + 6, w * v1.z);
    atomicAdd(ap + 7, w * v1.w);
}

__global__ __launch_bounds__(256) void k_relu_slice(
    float* __restrict__ p, int ld, size_t rows)
{
    size_t idx = (size_t)blockIdx.x * 256 + threadIdx.x;
    if (idx >= rows * NHID_) return;
    size_t r = idx >> 8;
    int    c = (int)(idx & 255);
    float* q = p + r * ld + c;
    float v = *q;
    *q = v > 0.0f ? v : 0.0f;
}

// ---------------------------------------------------------------------------
extern "C" void kernel_launch(void* const* d_in, const int* in_sizes, int n_in,
                              void* d_out, int out_size, void* d_ws, size_t ws_size,
                              hipStream_t stream)
{
    const float* x      = (const float*)d_in[0];
    const float* evo    = (const float*)d_in[1];
    const int*   esrc   = (const int*)  d_in[2];
    const int*   edst   = (const int*)  d_in[3];
    const float* ew     = (const float*)d_in[4];
    const float* conv_w = (const float*)d_in[5];
    const float* fc0_w  = (const float*)d_in[6];
    const float* fc0_b  = (const float*)d_in[7];
    const float* fc1_w  = (const float*)d_in[8];
    const float* fc1_b  = (const float*)d_in[9];
    const float* fc2_w  = (const float*)d_in[10];
    const float* fc2_b  = (const float*)d_in[11];
    const float* fc3_w  = (const float*)d_in[12];
    const float* fc3_b  = (const float*)d_in[13];

    const int E  = in_sizes[2];
    const int Nn = in_sizes[0] / NFEAT_;     // 50000

    // ---- workspace layout (256B-aligned regions) ----
    char*  ws  = (char*)d_ws;
    size_t off = 0;
    auto alloc = [&](size_t bytes) -> void* {
        void* p = ws + off;
        off = (off + bytes + 255) & ~(size_t)255;
        return p;
    };
    __bf16* fc0_t  = (__bf16*)alloc((size_t)NFEAT_ * NHID_ * 2);           // [256][512]
    __bf16* conv_t = (__bf16*)alloc((size_t)NLAY_ * NHID_ * NHID_ * 2);    // 4x[256][256]
    __bf16* fc1_t  = (__bf16*)alloc((size_t)EVO_ * NHID_ * 2);             // [256][1024]
    __bf16* fc2_t  = (__bf16*)alloc((size_t)NLAY_ * NHID_ * NHID_ * 2);    // [256][1024]
    __bf16* fc3_t  = (__bf16*)alloc((size_t)2 * NHID_ * NHID_ * 2);        // [256][512]
    float*  buf0   = (float*)alloc((size_t)Nn * NHID_ * 4);                // local after fc0
    float*  h      = (float*)alloc((size_t)Nn * NHID_ * 4);                // dense transform
    float*  cat    = (float*)alloc((size_t)Nn * NLAY_ * NHID_ * 4);        // layer concat / agg
    float*  cat2   = (float*)alloc((size_t)Nn * 2 * NHID_ * 4);            // [glob | local]

    // ---- weight conversion / transposition (bf16, N-major) ----
    k_transpose_bf16<<<(NFEAT_*NHID_ + 255)/256, 256, 0, stream>>>(fc0_w, fc0_t, NFEAT_, NHID_);
    for (int l = 0; l < NLAY_; ++l)
        k_transpose_bf16<<<(NHID_*NHID_ + 255)/256, 256, 0, stream>>>(
            conv_w + (size_t)l*NHID_*NHID_, conv_t + (size_t)l*NHID_*NHID_, NHID_, NHID_);
    k_transpose_bf16<<<(EVO_*NHID_ + 255)/256, 256, 0, stream>>>(fc1_w, fc1_t, EVO_, NHID_);
    k_transpose_bf16<<<(NLAY_*NHID_*NHID_ + 255)/256, 256, 0, stream>>>(fc2_w, fc2_t, NLAY_*NHID_, NHID_);
    k_transpose_bf16<<<(2*NHID_*NHID_ + 255)/256, 256, 0, stream>>>(fc3_w, fc3_t, 2*NHID_, NHID_);

    // ---- zero aggregation target (layer concat buffer) ----
    k_zero<<<2048, 256, 0, stream>>>(cat, (size_t)Nn * NLAY_ * NHID_);

    const int gblk = (Nn + 127) / 128;   // 128 rows per block (8 waves x 16)

    // fc0: local0 = relu(x @ W0 + b0)
    k_gemm_wmma<<<gblk, 256, 0, stream>>>(x, NFEAT_, fc0_t, fc0_b, buf0, NHID_, Nn, NFEAT_, 1);

    // 4 graph-conv layers
    const float* lin = buf0;
    int lda = NHID_;
    const int sblk = (E * 32 + 255) / 256;
    const int rblk = (int)(((size_t)Nn * NHID_ + 255) / 256);
    for (int l = 0; l < NLAY_; ++l) {
        k_gemm_wmma<<<gblk, 256, 0, stream>>>(lin, lda, conv_t + (size_t)l*NHID_*NHID_,
                                              nullptr, h, NHID_, Nn, NHID_, 0);
        k_spmm_scatter<<<sblk, 256, 0, stream>>>(h, esrc, edst, ew,
                                                 cat + (size_t)l*NHID_, NLAY_*NHID_, E);
        k_relu_slice<<<rblk, 256, 0, stream>>>(cat + (size_t)l*NHID_, NLAY_*NHID_, (size_t)Nn);
        lin = cat + (size_t)l*NHID_;
        lda = NLAY_*NHID_;
    }

    // fc2: local = relu(concat(layers) @ W2 + b2) -> right half of cat2
    k_gemm_wmma<<<gblk, 256, 0, stream>>>(cat, NLAY_*NHID_, fc2_t, fc2_b,
                                          cat2 + NHID_, 2*NHID_, Nn, NLAY_*NHID_, 1);
    // fc1: glob = relu(evo @ W1 + b1) -> left half of cat2
    k_gemm_wmma<<<gblk, 256, 0, stream>>>(evo, EVO_, fc1_t, fc1_b,
                                          cat2, 2*NHID_, Nn, EVO_, 1);
    // fc3: out = relu([glob|local] @ W3 + b3)
    k_gemm_wmma<<<gblk, 256, 0, stream>>>(cat2, 2*NHID_, fc3_t, fc3_b,
                                          (float*)d_out, NHID_, Nn, 2*NHID_, 1);
}
